// TurboQuantKVCache_85942295593389
// MI455X (gfx1250) — matrix operand order repaired and checked by reading, not compiled, not run on hardware
//
#include <hip/hip_runtime.h>
#include <stdint.h>

typedef __bf16 v16bf __attribute__((ext_vector_type(16)));
typedef float  v8f   __attribute__((ext_vector_type(8)));

#define N_HEADS   16
#define HEAD_DIM  128
#define HALF_DIM  64
#define MAX_SEQ   16384
#define T_SEQ     4096

// d_out layout in float elements (tuple concatenated flat: kp, kn, vp, vn)
#define KP_OFF 0UL
#define KN_OFF (16UL * 16384UL * 64UL)              // 16,777,216
#define VP_OFF (KN_OFF + 16UL * 16384UL)            // 17,039,360
#define VN_OFF (VP_OFF + 16UL * 16384UL * 64UL)     // 33,816,576
#define OUT_TOTAL (VN_OFF + 16UL * 16384UL)         // 34,078,720

#define WAVES_PER_BLOCK 4
#define THREADS (WAVES_PER_BLOCK * 32)
#define TILES_PER_WAVE 2
#define TILES_PER_BLOCK (WAVES_PER_BLOCK * TILES_PER_WAVE)
#define TOTAL_TILES (2 * N_HEADS * (T_SEQ / 16))    // 8192
#define NUM_BLOCKS (TOTAL_TILES / TILES_PER_BLOCK)  // 1024

// fragment buffer: 1024 lane-fragments, 48B stride (32B payload + 16B pad
// so lanes l and l+8 hit different LDS banks on the b128 reads)
#define FRAG_U4_STRIDE 3          // 3 * 16B

union BFrag { v16bf v; uint4 u[2]; };

__device__ __forceinline__ v8f wmma_bf16(v16bf a, v16bf b, v8f c) {
    return __builtin_amdgcn_wmma_f32_16x16x32_bf16(false, a, false, b,
                                                   (short)0, c, false, false);
}

__global__ __launch_bounds__(THREADS)
void tq_kernel(const int* __restrict__ input_pos,
               const float* __restrict__ k_val,
               const float* __restrict__ v_val,
               const __bf16* __restrict__ rot,      // rotation_T, 128x128 row-major
               const __bf16* __restrict__ bnd_bf,   // 15 boundaries
               float* __restrict__ out)
{
    __shared__ __bf16 srot[HEAD_DIM * HEAD_DIM];        // 32 KB raw copy
    __shared__ uint4  sfrag4[1024 * FRAG_U4_STRIDE];    // 48 KB fragment-ordered

    const int tid  = threadIdx.x;
    const int wave = tid >> 5;
    const int lane = tid & 31;
    const int h    = lane >> 4;   // half select
    const int nl   = lane & 15;

    // ---- stage rotation_T into LDS via CDNA5 async-to-LDS (ASYNCcnt path) ----
    {
        const uint64_t gbase = (uint64_t)(uintptr_t)rot;
        const uint32_t lbase = (uint32_t)(uintptr_t)(&srot[0]);
        #pragma unroll
        for (int i = 0; i < 16; ++i) {   // 128 thr * 16 iters * 16B = 32 KB
            uint32_t off   = (uint32_t)(i * THREADS + tid) * 16u;
            uint32_t laddr = lbase + off;
            uint64_t gaddr = gbase + off;
            asm volatile("global_load_async_to_lds_b128 %0, %1, off"
                         :: "v"(laddr), "v"(gaddr) : "memory");
        }
        asm volatile("s_wait_asynccnt 0" ::: "memory");
    }
    __syncthreads();

    // ---- one-time reorg: rotation -> B-fragment order ----
    // lane-fragment f = ((nt*4 + c)*32 + lane); 16 halves contiguous per lane.
    #pragma unroll 1
    for (int i = 0; i < 8; ++i) {
        const int f    = tid + THREADS * i;   // 0..1023
        const int lf   = f & 31;
        const int cidx = (f >> 5) & 3;
        const int ntix = f >> 7;
        const int hf   = lf >> 4;
        const int nlf  = lf & 15;
        BFrag bb;
        #pragma unroll
        for (int e = 0; e < 16; ++e)
            bb.v[e] = srot[(32*cidx + 8*hf + (e & 7) + 16*(e >> 3)) * HEAD_DIM + 16*ntix + nlf];
        sfrag4[f * FRAG_U4_STRIDE + 0] = bb.u[0];
        sfrag4[f * FRAG_U4_STRIDE + 1] = bb.u[1];
    }
    __syncthreads();

    // boundaries (uniform, bf16 -> f32; stays in SGPRs)
    float bnd[15];
    #pragma unroll
    for (int j = 0; j < 15; ++j) bnd[j] = (float)bnd_bf[j];

    // ---- this wave's two consecutive tiles (same tensor+head) ----
    const int tile0 = blockIdx.x * TILES_PER_BLOCK + wave * TILES_PER_WAVE;
    const int sel   = tile0 >> 12;            // 0 = K tensor, 1 = V tensor
    const int head  = (tile0 >> 8) & 15;

    const float* src      = sel ? v_val : k_val;
    float*       pack_out = out + (sel ? VP_OFF : KP_OFF);
    float*       norm_out = out + (sel ? VN_OFF : KN_OFF);

    v16bf a[TILES_PER_WAVE][4];
    int   ipos[TILES_PER_WAVE][8];

    #pragma unroll
    for (int tt = 0; tt < TILES_PER_WAVE; ++tt) {
        const int row0 = ((tile0 & 255) + tt) * 16;

        // load 16x128 tile: this lane covers row nl, K-half h, in A-fragment order
        const float* rowp = src + (size_t)(head * T_SEQ + row0 + nl) * HEAD_DIM;
        float v[64];
        #pragma unroll
        for (int c = 0; c < 4; ++c) {
            const int K0 = 32 * c + 8 * h;
            const float4 f0 = *(const float4*)(rowp + K0);
            const float4 f1 = *(const float4*)(rowp + K0 + 4);
            const float4 f2 = *(const float4*)(rowp + K0 + 16);
            const float4 f3 = *(const float4*)(rowp + K0 + 20);
            v[c*16+ 0]=f0.x; v[c*16+ 1]=f0.y; v[c*16+ 2]=f0.z; v[c*16+ 3]=f0.w;
            v[c*16+ 4]=f1.x; v[c*16+ 5]=f1.y; v[c*16+ 6]=f1.z; v[c*16+ 7]=f1.w;
            v[c*16+ 8]=f2.x; v[c*16+ 9]=f2.y; v[c*16+10]=f2.z; v[c*16+11]=f2.w;
            v[c*16+12]=f3.x; v[c*16+13]=f3.y; v[c*16+14]=f3.z; v[c*16+15]=f3.w;
        }

        // per-row L2 norm: lane pair (l, l^16) covers the full row
        float s = 0.f;
        #pragma unroll
        for (int i = 0; i < 64; ++i) s += v[i] * v[i];
        s += __shfl_xor(s, 16, 32);
        const float  normf = sqrtf(s);
        const __bf16 nb    = (__bf16)normf;              // bf16 rounding as reference
        const float  scale = 1.0f / ((float)nb + 1e-10f);

        #pragma unroll
        for (int c = 0; c < 4; ++c)
            #pragma unroll
            for (int e = 0; e < 16; ++e)
                a[tt][c][e] = (__bf16)((float)(__bf16)v[c*16 + e] * scale);

        #pragma unroll
        for (int i = 0; i < 8; ++i) ipos[tt][i] = input_pos[row0 + 8*h + i];

        if (h == 0)
            norm_out[head * MAX_SEQ + input_pos[row0 + nl]] = (float)nb;
    }

    // ---- rotate (WMMA, B shared across both tiles) + quantize + pack ----
    #pragma unroll 1
    for (int nt = 0; nt < 8; ++nt) {
        v8f acc0 = {}, acc1 = {};
        #pragma unroll
        for (int c = 0; c < 4; ++c) {
            const int fi = (nt * 4 + c) * 32 + lane;
            BFrag bb;
            bb.u[0] = sfrag4[fi * FRAG_U4_STRIDE + 0];   // ds_load_b128
            bb.u[1] = sfrag4[fi * FRAG_U4_STRIDE + 1];   // ds_load_b128
            acc0 = wmma_bf16(a[0][c], bb.v, acc0);
            acc1 = wmma_bf16(a[1][c], bb.v, acc1);
        }
        #pragma unroll
        for (int tt = 0; tt < TILES_PER_WAVE; ++tt) {
            const v8f acc = tt ? acc1 : acc0;
            #pragma unroll
            for (int i = 0; i < 8; ++i) {
                // reference casts bf16 GEMM result to f32 before bucketize
                const float x = (float)(__bf16)acc[i];
                int idx = 0;
                #pragma unroll
                for (int j = 0; j < 15; ++j) idx += (bnd[j] < x) ? 1 : 0; // side='left'
                const int other = __shfl_xor(idx, 1, 32);   // partner column N^1
                if ((lane & 1) == 0) {                      // even column owns the byte
                    const int byte = (idx << 4) | other;    // even col -> high nibble
                    pack_out[((size_t)head * MAX_SEQ + ipos[tt][i]) * HALF_DIM
                             + 8 * nt + (nl >> 1)] = (float)byte;
                }
            }
        }
    }
}

extern "C" void kernel_launch(void* const* d_in, const int* in_sizes, int n_in,
                              void* d_out, int out_size, void* d_ws, size_t ws_size,
                              hipStream_t stream) {
    (void)in_sizes; (void)n_in; (void)d_ws; (void)ws_size;
    const int*    input_pos = (const int*)d_in[0];
    const float*  k_val     = (const float*)d_in[1];
    const float*  v_val     = (const float*)d_in[2];
    const __bf16* rot       = (const __bf16*)d_in[3];
    const __bf16* bnd       = (const __bf16*)d_in[4];
    float*        out       = (float*)d_out;

    // zero the full cache (rows 4096..16383 must be zero; harness poisons once)
    size_t bytes = (size_t)out_size * sizeof(float);
    size_t want  = OUT_TOTAL * sizeof(float);
    hipMemsetAsync(d_out, 0, bytes < want ? bytes : want, stream);

    tq_kernel<<<NUM_BLOCKS, THREADS, 0, stream>>>(input_pos, k_val, v_val, rot, bnd, out);
}
